// MEGNN_81750407512687
// MI455X (gfx1250) — matrix operand order repaired and checked by reading, not compiled, not run on hardware
//
#include <hip/hip_runtime.h>

typedef __attribute__((ext_vector_type(16))) _Float16 v16h;
typedef __attribute__((ext_vector_type(8)))  float    v8f;
typedef _Float16 h16_t;

#define G_  2
#define L_  7
#define H_  128
#define IN_ 15
#define ED_ 4
#define N_  8192
#define E_  253952
#define NN_ 32
#define B_  256
#define KE  261   // 2H + 1 + ED
#define KN  271   // 2H + IN

union FragU { v16h v; unsigned u[8]; };

__device__ __forceinline__ float silu_f(float x)    { return x / (1.0f + __expf(-x)); }
__device__ __forceinline__ float sigmoid_f(float x) { return 1.0f / (1.0f + __expf(-x)); }

// ---------------------------------------------------------------- prep kernels

__global__ void zero_kernel(float* p, long n) {
    long i = (long)blockIdx.x * blockDim.x + threadIdx.x;
    if (i < n) p[i] = 0.f;
}

// src: [pairs][srcK][H_] f32 row-major; dst: [pairs][H_][Kd] f16 (k<Kd only)
__global__ void transpose_w_kernel(const float* __restrict__ src, h16_t* __restrict__ dst,
                                   int pairs, int srcK, int Kd) {
    long i = (long)blockIdx.x * blockDim.x + threadIdx.x;
    long tot = (long)pairs * H_ * Kd;
    if (i >= tot) return;
    int  k  = (int)(i % Kd);
    long r  = i / Kd;
    int  n  = (int)(r % H_);
    long pr = r / H_;
    dst[i] = (h16_t)src[(pr * srcK + k) * H_ + n];
}

__global__ void embed_kernel(const float* __restrict__ h0, const float* __restrict__ emb_w,
                             const float* __restrict__ emb_b,
                             float* __restrict__ hbuf, h16_t* __restrict__ h16) {
    long i = (long)blockIdx.x * blockDim.x + threadIdx.x;
    if (i >= (long)G_ * N_ * H_) return;
    int  c  = (int)(i % H_);
    long gn = i / H_;
    const float* row = h0 + gn * IN_;
    float acc = emb_b[c];
#pragma unroll
    for (int t = 0; t < IN_; ++t) acc += row[t] * emb_w[t * H_ + c];
    hbuf[i] = acc;
    h16[i]  = (h16_t)acc;
}

__global__ void radial_kernel(const float* __restrict__ x, const int* __restrict__ edges,
                              float* __restrict__ radial) {
    long i = (long)blockIdx.x * blockDim.x + threadIdx.x;
    if (i >= (long)G_ * E_) return;
    int g = (int)(i / E_), e = (int)(i % E_);
    int r = edges[(g * 2 + 0) * E_ + e];
    int c = edges[(g * 2 + 1) * E_ + e];
    const float* xr = x + ((long)g * N_ + r) * 3;
    const float* xc = x + ((long)g * N_ + c) * 3;
    float dx = xr[0] - xc[0], dy = xr[1] - xc[1], dz = xr[2] - xc[2];
    radial[i] = dx * dx + dy * dy + dz * dz;
}

// ---------------------------------------------------------------- edge model
// One wave handles 16 edges x 128 outputs. Block = 4 waves = 64 edges.

struct EdgeLdsWave {
    h16_t A1[16 * 264];   // concat(h[row],h[col]) f16, stride 264 (even)
    h16_t M1[16 * 128];   // silu(GEMM1) staged for GEMM2 A-fragments
    float tail[16 * 8];   // radial, eattr[4], emask
    int   rown[16];
    int   coln[16];
};

__launch_bounds__(128, 1)
__global__ void egcl_edge_kernel(const h16_t* __restrict__ h16,
                                 const float* __restrict__ radial,
                                 const int*   __restrict__ edges,
                                 const float* __restrict__ eattr,
                                 const float* __restrict__ emaskg,
                                 const h16_t* __restrict__ ew1T,   // [G*L][128][256] f16
                                 const float* __restrict__ ew1f,   // [G*L][261][128] f32
                                 const float* __restrict__ eb1,
                                 const h16_t* __restrict__ ew2T,   // [G*L][128][128] f16
                                 const float* __restrict__ eb2,
                                 const float* __restrict__ aw,
                                 const float* __restrict__ ab,
                                 float* __restrict__ agg,
                                 int li) {
    __shared__ EdgeLdsWave lds[4];
    const int g    = blockIdx.y;
    const int wave = threadIdx.x >> 5;
    const int lane = threadIdx.x & 31;
    EdgeLdsWave& W = lds[wave];
    const int e0 = (blockIdx.x * 4 + wave) * 16;

    if (lane < 16) {
        int e = e0 + lane;
        W.rown[lane] = edges[(g * 2 + 0) * E_ + e];
        W.coln[lane] = edges[(g * 2 + 1) * E_ + e];
        W.tail[lane * 8 + 0] = radial[(long)g * E_ + e];
#pragma unroll
        for (int d = 0; d < ED_; ++d)
            W.tail[lane * 8 + 1 + d] = eattr[((long)g * E_ + e) * ED_ + d];
        W.tail[lane * 8 + 5] = emaskg[(long)g * E_ + e];
    }
    __syncthreads();

    // gather h16[row] ++ h16[col] into A1 (dword-pair loads)
    const unsigned* h16u = (const unsigned*)h16;
    unsigned* A1u = (unsigned*)W.A1;
#pragma unroll 4
    for (int it = 0; it < 64; ++it) {
        int idx = it * 32 + lane;        // 16 edges x 128 dwords
        int e = idx >> 7, p = idx & 127;
        int node = (p < 64) ? W.rown[e] : W.coln[e];
        A1u[e * 132 + p] = h16u[((long)g * N_ + node) * 64 + (p & 63)];
    }
    __syncthreads();

    const int nl  = lane & 15;
    const int mg  = (lane >> 4) * 8;   // C fragment row base
    const int kbA = (lane >> 4) * 8;   // A fragment k offset
    const int kbB = (lane >> 4) * 16;  // unused; B fragments use kbA
    const long wl = (long)(g * L_ + li);

    // ---- GEMM1: [16x261]@[261x128]; K=256 via WMMA, tail (radial+eattr) in f32
    v8f acc[8];
#pragma unroll
    for (int nt = 0; nt < 8; ++nt) {
        int nc = nt * 16 + nl;
        float b    = eb1[wl * H_ + nc];
        float w256 = ew1f[(wl * KE + 256) * H_ + nc];
        float w257 = ew1f[(wl * KE + 257) * H_ + nc];
        float w258 = ew1f[(wl * KE + 258) * H_ + nc];
        float w259 = ew1f[(wl * KE + 259) * H_ + nc];
        float w260 = ew1f[(wl * KE + 260) * H_ + nc];
#pragma unroll
        for (int v = 0; v < 8; ++v) {
            const float* t = &W.tail[(mg + v) * 8];
            acc[nt][v] = b + t[0] * w256 + t[1] * w257 + t[2] * w258 + t[3] * w259 + t[4] * w260;
        }
    }
    const unsigned* Au  = (const unsigned*)W.A1;
    const unsigned* B1u = (const unsigned*)(ew1T + wl * H_ * 256);
#pragma unroll
    for (int ks = 0; ks < 8; ++ks) {
        FragU af;
#pragma unroll
        for (int v = 0; v < 8; ++v) {
            int k = ks * 32 + ((v >> 2) << 4) + kbA + ((v & 3) << 1);
            af.u[v] = Au[((lane & 15) * 264 + k) >> 1];
        }
#pragma unroll
        for (int nt = 0; nt < 8; ++nt) {
            int nc = nt * 16 + nl;
            FragU bf;
#pragma unroll
            for (int v = 0; v < 8; ++v) {
                int k = ks * 32 + ((v >> 2) << 4) + kbA + ((v & 3) << 1);
                bf.u[v] = B1u[(nc * 256 + k) >> 1];
            }
            acc[nt] = __builtin_amdgcn_wmma_f32_16x16x32_f16(false, af.v, false, bf.v,
                                                             (short)0, acc[nt], false, false);
        }
    }
    // silu -> M1 (f16)
#pragma unroll
    for (int nt = 0; nt < 8; ++nt)
#pragma unroll
        for (int v = 0; v < 8; ++v)
            W.M1[(mg + v) * 128 + nt * 16 + nl] = (h16_t)silu_f(acc[nt][v]);
    __syncthreads();

    // ---- GEMM2: [16x128]@[128x128] + silu
    v8f acc2[8];
#pragma unroll
    for (int nt = 0; nt < 8; ++nt) {
        float b = eb2[wl * H_ + nt * 16 + nl];
#pragma unroll
        for (int v = 0; v < 8; ++v) acc2[nt][v] = b;
    }
    const unsigned* M1u = (const unsigned*)W.M1;
    const unsigned* B2u = (const unsigned*)(ew2T + wl * H_ * 128);
#pragma unroll
    for (int ks = 0; ks < 4; ++ks) {
        FragU af;
#pragma unroll
        for (int v = 0; v < 8; ++v) {
            int k = ks * 32 + ((v >> 2) << 4) + kbA + ((v & 3) << 1);
            af.u[v] = M1u[((lane & 15) * 128 + k) >> 1];
        }
#pragma unroll
        for (int nt = 0; nt < 8; ++nt) {
            int nc = nt * 16 + nl;
            FragU bf;
#pragma unroll
            for (int v = 0; v < 8; ++v) {
                int k = ks * 32 + ((v >> 2) << 4) + kbA + ((v & 3) << 1);
                bf.u[v] = B2u[(nc * 128 + k) >> 1];
            }
            acc2[nt] = __builtin_amdgcn_wmma_f32_16x16x32_f16(false, af.v, false, bf.v,
                                                              (short)0, acc2[nt], false, false);
        }
    }
#pragma unroll
    for (int nt = 0; nt < 8; ++nt)
#pragma unroll
        for (int v = 0; v < 8; ++v) acc2[nt][v] = silu_f(acc2[nt][v]);

    // ---- attention gate + edge mask + scatter-add to agg[row]
    float awv[8];
#pragma unroll
    for (int nt = 0; nt < 8; ++nt) awv[nt] = aw[wl * H_ + nt * 16 + nl];
    const float abv = ab[wl];
#pragma unroll
    for (int v = 0; v < 8; ++v) {
        float s = 0.f;
#pragma unroll
        for (int nt = 0; nt < 8; ++nt) s += acc2[nt][v] * awv[nt];
        s += __shfl_xor(s, 1, 16);
        s += __shfl_xor(s, 2, 16);
        s += __shfl_xor(s, 4, 16);
        s += __shfl_xor(s, 8, 16);
        float gate = sigmoid_f(s + abv) * W.tail[(mg + v) * 8 + 5];
        int r = W.rown[mg + v];
        float* dst = agg + ((long)g * N_ + r) * H_ + nl;
#pragma unroll
        for (int nt = 0; nt < 8; ++nt)
            unsafeAtomicAdd(dst + nt * 16, acc2[nt][v] * gate);
    }
}

// ---------------------------------------------------------------- node model
// One wave handles 16 nodes x 128 outputs; also resets agg to 0 for next layer.

struct NodeLdsWave {
    h16_t A1[16 * 256];   // concat(h, agg) f16
    h16_t M1[16 * 128];
    float attr[16 * 16];  // node_attr (15 used)
};

__launch_bounds__(128, 1)
__global__ void egcl_node_kernel(float* __restrict__ hbuf,
                                 h16_t* __restrict__ h16,
                                 float* __restrict__ agg,
                                 const float* __restrict__ h0,
                                 const h16_t* __restrict__ nw1T,   // [G*L][128][256] f16
                                 const float* __restrict__ nw1f,   // [G*L][271][128] f32
                                 const float* __restrict__ nb1,
                                 const h16_t* __restrict__ nw2T,   // [G*L][128][128] f16
                                 const float* __restrict__ nb2,
                                 int li) {
    __shared__ NodeLdsWave lds[4];
    const int g    = blockIdx.y;
    const int wave = threadIdx.x >> 5;
    const int lane = threadIdx.x & 31;
    NodeLdsWave& W = lds[wave];
    const int n0 = (blockIdx.x * 4 + wave) * 16;

    const unsigned* h16u = (const unsigned*)h16;
    unsigned* A1u = (unsigned*)W.A1;
#pragma unroll 4
    for (int it = 0; it < 32; ++it) {            // h -> A1[:,0:128]
        int idx = it * 32 + lane;                // 16 rows x 64 dwords
        int m = idx >> 6, p = idx & 63;
        A1u[m * 128 + p] = h16u[((long)g * N_ + n0 + m) * 64 + p];
    }
#pragma unroll 4
    for (int it = 0; it < 64; ++it) {            // agg -> A1[:,128:256] (f32->f16)
        int idx = it * 32 + lane;                // 16 rows x 128 floats
        int m = idx >> 7, p = idx & 127;
        W.A1[m * 256 + 128 + p] = (h16_t)agg[((long)g * N_ + n0 + m) * H_ + p];
    }
#pragma unroll
    for (int it = 0; it < 8; ++it) {             // node_attr (h0)
        int idx = it * 32 + lane;                // 16 rows x 16 slots
        int m = idx >> 4, p = idx & 15;
        W.attr[m * 16 + p] = (p < IN_) ? h0[((long)g * N_ + n0 + m) * IN_ + p] : 0.f;
    }
    __syncthreads();

    const int nl  = lane & 15;
    const int mg  = (lane >> 4) * 8;
    const int kbA = (lane >> 4) * 8;
    const int kbB = (lane >> 4) * 16;
    const long wl = (long)(g * L_ + li);

    // ---- GEMM1: [16x271]@[271x128]; K=256 WMMA, node_attr tail in f32
    v8f acc[8];
#pragma unroll
    for (int nt = 0; nt < 8; ++nt) {
        int nc = nt * 16 + nl;
        float wt[IN_];
#pragma unroll
        for (int t = 0; t < IN_; ++t) wt[t] = nw1f[(wl * KN + 256 + t) * H_ + nc];
        float b = nb1[wl * H_ + nc];
#pragma unroll
        for (int v = 0; v < 8; ++v) {
            float a = b;
            const float* at = &W.attr[(mg + v) * 16];
#pragma unroll
            for (int t = 0; t < IN_; ++t) a += at[t] * wt[t];
            acc[nt][v] = a;
        }
    }
    const unsigned* Au  = (const unsigned*)W.A1;
    const unsigned* B1u = (const unsigned*)(nw1T + wl * H_ * 256);
#pragma unroll
    for (int ks = 0; ks < 8; ++ks) {
        FragU af;
#pragma unroll
        for (int v = 0; v < 8; ++v) {
            int k = ks * 32 + ((v >> 2) << 4) + kbA + ((v & 3) << 1);
            af.u[v] = Au[((lane & 15) * 256 + k) >> 1];
        }
#pragma unroll
        for (int nt = 0; nt < 8; ++nt) {
            int nc = nt * 16 + nl;
            FragU bf;
#pragma unroll
            for (int v = 0; v < 8; ++v) {
                int k = ks * 32 + ((v >> 2) << 4) + kbA + ((v & 3) << 1);
                bf.u[v] = B1u[(nc * 256 + k) >> 1];
            }
            acc[nt] = __builtin_amdgcn_wmma_f32_16x16x32_f16(false, af.v, false, bf.v,
                                                             (short)0, acc[nt], false, false);
        }
    }
#pragma unroll
    for (int nt = 0; nt < 8; ++nt)
#pragma unroll
        for (int v = 0; v < 8; ++v)
            W.M1[(mg + v) * 128 + nt * 16 + nl] = (h16_t)silu_f(acc[nt][v]);
    __syncthreads();

    // ---- GEMM2: [16x128]@[128x128] + bias (no silu), then recurrent add
    v8f acc2[8];
#pragma unroll
    for (int nt = 0; nt < 8; ++nt) {
        float b = nb2[wl * H_ + nt * 16 + nl];
#pragma unroll
        for (int v = 0; v < 8; ++v) acc2[nt][v] = b;
    }
    const unsigned* M1u = (const unsigned*)W.M1;
    const unsigned* B2u = (const unsigned*)(nw2T + wl * H_ * 128);
#pragma unroll
    for (int ks = 0; ks < 4; ++ks) {
        FragU af;
#pragma unroll
        for (int v = 0; v < 8; ++v) {
            int k = ks * 32 + ((v >> 2) << 4) + kbA + ((v & 3) << 1);
            af.u[v] = M1u[((lane & 15) * 128 + k) >> 1];
        }
#pragma unroll
        for (int nt = 0; nt < 8; ++nt) {
            int nc = nt * 16 + nl;
            FragU bf;
#pragma unroll
            for (int v = 0; v < 8; ++v) {
                int k = ks * 32 + ((v >> 2) << 4) + kbA + ((v & 3) << 1);
                bf.u[v] = B2u[(nc * 128 + k) >> 1];
            }
            acc2[nt] = __builtin_amdgcn_wmma_f32_16x16x32_f16(false, af.v, false, bf.v,
                                                              (short)0, acc2[nt], false, false);
        }
    }
    // h = h + delta; refresh f16 mirror; zero agg for next layer
#pragma unroll
    for (int v = 0; v < 8; ++v) {
        long base = ((long)g * N_ + n0 + mg + v) * H_;
#pragma unroll
        for (int nt = 0; nt < 8; ++nt) {
            int c = nt * 16 + nl;
            float hv = hbuf[base + c] + acc2[nt][v];
            hbuf[base + c] = hv;
            h16[base + c]  = (h16_t)hv;
            agg[base + c]  = 0.f;
        }
    }
}

// ---------------------------------------------------------------- decoders

__global__ void dec1_kernel(const float* __restrict__ hbuf, const float* __restrict__ dw1,
                            const float* __restrict__ db1, float* __restrict__ t) {
    long i = (long)blockIdx.x * blockDim.x + threadIdx.x;
    if (i >= (long)G_ * N_ * H_) return;
    int  c  = (int)(i % H_);
    long gn = i / H_;
    int  g  = (int)(gn / N_);
    const float* row = hbuf + gn * H_;
    const float* w   = dw1 + (long)g * H_ * H_;
    float acc = db1[g * H_ + c];
    for (int k = 0; k < H_; ++k) acc += row[k] * w[k * H_ + c];
    t[i] = silu_f(acc);
}

__global__ void dec2_kernel(const float* __restrict__ t, const float* __restrict__ dw2,
                            const float* __restrict__ db2, const float* __restrict__ nmask,
                            float* __restrict__ pooled) {
    long i = (long)blockIdx.x * blockDim.x + threadIdx.x;
    if (i >= (long)G_ * N_ * H_) return;
    int  c  = (int)(i % H_);
    long gn = i / H_;
    int  g  = (int)(gn / N_);
    int  n  = (int)(gn % N_);
    const float* row = t + gn * H_;
    const float* w   = dw2 + (long)g * H_ * H_;
    float acc = db2[g * H_ + c];
    for (int k = 0; k < H_; ++k) acc += row[k] * w[k * H_ + c];
    acc *= nmask[gn];
    unsafeAtomicAdd(&pooled[((long)g * B_ + (n / NN_)) * H_ + c], acc);
}

__global__ void grand1_kernel(const float* __restrict__ pooled, const float* __restrict__ gw1,
                              const float* __restrict__ gb1, float* __restrict__ t2) {
    int i = blockIdx.x * blockDim.x + threadIdx.x;
    if (i >= B_ * G_ * H_) return;
    int c = i % (G_ * H_), b = i / (G_ * H_);
    float acc = gb1[c];
    for (int k = 0; k < G_ * H_; ++k) {
        float cv = pooled[((long)(k / H_) * B_ + b) * H_ + (k % H_)];
        acc += cv * gw1[k * (G_ * H_) + c];
    }
    t2[b * (G_ * H_) + c] = silu_f(acc);
}

__global__ void grand2_kernel(const float* __restrict__ t2, const float* __restrict__ gw2,
                              const float* __restrict__ gb2, float* __restrict__ out) {
    int b = blockIdx.x * blockDim.x + threadIdx.x;
    if (b >= B_) return;
    float acc = gb2[0];
    for (int k = 0; k < G_ * H_; ++k) acc += t2[b * (G_ * H_) + k] * gw2[k];
    out[b] = acc;
}

// ---------------------------------------------------------------- launch

extern "C" void kernel_launch(void* const* d_in, const int* in_sizes, int n_in,
                              void* d_out, int out_size, void* d_ws, size_t ws_size,
                              hipStream_t stream) {
    (void)in_sizes; (void)n_in; (void)out_size; (void)ws_size;
    const float* h0    = (const float*)d_in[0];
    const float* x     = (const float*)d_in[1];
    const int*   edges = (const int*)  d_in[2];
    const float* eattr = (const float*)d_in[3];
    const float* nmask = (const float*)d_in[4];
    const float* emask = (const float*)d_in[5];
    // d_in[6] = n_nodes (== 32, compile-time NN_)
    const float* emb_w = (const float*)d_in[7];
    const float* emb_b = (const float*)d_in[8];
    const float* ew1   = (const float*)d_in[9];
    const float* eb1   = (const float*)d_in[10];
    const float* ew2   = (const float*)d_in[11];
    const float* eb2   = (const float*)d_in[12];
    const float* aw    = (const float*)d_in[13];
    const float* ab    = (const float*)d_in[14];
    const float* nw1   = (const float*)d_in[15];
    const float* nb1   = (const float*)d_in[16];
    const float* nw2   = (const float*)d_in[17];
    const float* nb2   = (const float*)d_in[18];
    const float* dw1   = (const float*)d_in[19];
    const float* db1   = (const float*)d_in[20];
    const float* dw2   = (const float*)d_in[21];
    const float* db2   = (const float*)d_in[22];
    const float* gw1   = (const float*)d_in[23];
    const float* gb1   = (const float*)d_in[24];
    const float* gw2   = (const float*)d_in[25];
    const float* gb2   = (const float*)d_in[26];
    float* out = (float*)d_out;

    char* ws = (char*)d_ws;
    size_t off = 0;
    auto alloc = [&](size_t bytes) -> void* {
        void* p = ws + off;
        off += (bytes + 255) & ~(size_t)255;
        return p;
    };
    float* hbuf   = (float*)alloc((size_t)G_ * N_ * H_ * 4);
    h16_t* h16    = (h16_t*)alloc((size_t)G_ * N_ * H_ * 2);
    float* agg    = (float*)alloc((size_t)G_ * N_ * H_ * 4);   // reused as decoder temp
    float* radial = (float*)alloc((size_t)G_ * E_ * 4);
    h16_t* ew1T   = (h16_t*)alloc((size_t)G_ * L_ * H_ * 256 * 2);
    h16_t* ew2T   = (h16_t*)alloc((size_t)G_ * L_ * H_ * 128 * 2);
    h16_t* nw1T   = (h16_t*)alloc((size_t)G_ * L_ * H_ * 256 * 2);
    h16_t* nw2T   = (h16_t*)alloc((size_t)G_ * L_ * H_ * 128 * 2);
    float* pooled = (float*)alloc((size_t)G_ * B_ * H_ * 4);
    float* t2     = (float*)alloc((size_t)B_ * G_ * H_ * 4);

    { long n = (long)G_ * N_ * H_; zero_kernel<<<(n + 255) / 256, 256, 0, stream>>>(agg, n); }
    { long n = (long)G_ * B_ * H_; zero_kernel<<<(n + 255) / 256, 256, 0, stream>>>(pooled, n); }

    { long n = (long)G_ * L_ * H_ * 256;
      transpose_w_kernel<<<(n + 255) / 256, 256, 0, stream>>>(ew1, ew1T, G_ * L_, KE, 256); }
    { long n = (long)G_ * L_ * H_ * 128;
      transpose_w_kernel<<<(n + 255) / 256, 256, 0, stream>>>(ew2, ew2T, G_ * L_, 128, 128); }
    { long n = (long)G_ * L_ * H_ * 256;
      transpose_w_kernel<<<(n + 255) / 256, 256, 0, stream>>>(nw1, nw1T, G_ * L_, KN, 256); }
    { long n = (long)G_ * L_ * H_ * 128;
      transpose_w_kernel<<<(n + 255) / 256, 256, 0, stream>>>(nw2, nw2T, G_ * L_, 128, 128); }

    { long n = (long)G_ * N_ * H_;
      embed_kernel<<<(n + 255) / 256, 256, 0, stream>>>(h0, emb_w, emb_b, hbuf, h16); }
    { long n = (long)G_ * E_;
      radial_kernel<<<(n + 255) / 256, 256, 0, stream>>>(x, edges, radial); }

    for (int li = 0; li < L_; ++li) {
        egcl_edge_kernel<<<dim3(E_ / 64, G_), 128, 0, stream>>>(
            h16, radial, edges, eattr, emask, ew1T, ew1, eb1, ew2T, eb2, aw, ab, agg, li);
        egcl_node_kernel<<<dim3(N_ / 64, G_), 128, 0, stream>>>(
            hbuf, h16, agg, h0, nw1T, nw1, nb1, nw2T, nb2, li);
    }

    { long n = (long)G_ * N_ * H_;
      dec1_kernel<<<(n + 255) / 256, 256, 0, stream>>>(hbuf, dw1, db1, agg); }
    { long n = (long)G_ * N_ * H_;
      dec2_kernel<<<(n + 255) / 256, 256, 0, stream>>>(agg, dw2, db2, nmask, pooled); }
    { int n = B_ * G_ * H_;
      grand1_kernel<<<(n + 255) / 256, 256, 0, stream>>>(pooled, gw1, gb1, t2); }
    grand2_kernel<<<1, 256, 0, stream>>>(t2, gw2, gb2, out);
}
